// WeightedNegInnerIteration_24507083391172
// MI455X (gfx1250) — compile-verified, matrix operand order checked by reading, not run on hardware
//
#include <hip/hip_runtime.h>
#include <hip/hip_bf16.h>
#include <math.h>

#define D   64
#define GD  8
#define NV  8192
#define NC  32768
#define NSEG 4   // K-split for GEMM2

typedef __bf16 bf16_t;
typedef __attribute__((ext_vector_type(16))) __bf16 v16bf;
typedef __attribute__((ext_vector_type(8)))  __bf16 v8bf;
typedef __attribute__((ext_vector_type(8)))  float  v8f;
typedef __attribute__((ext_vector_type(4)))  unsigned int v4u;

// Fragment layout helpers for 16x16x32 bf16 WMMA operands (A and B symmetric):
// lane = idx16 + 16*((kk>>3)&1), halfword h = (kk&7) + 8*((kk>>4)&1)
__device__ __forceinline__ int frag_lane(int kk, int idx16) {
  return idx16 + 16 * ((kk >> 3) & 1);
}
__device__ __forceinline__ int frag_half(int kk) {
  return (kk & 7) + 8 * ((kk >> 4) & 1);
}

// ---------------------------------------------------------------------------
// K1: vars_fwd = v @ fb0^T, written in WMMA B-fragment layout (bf16), 1 MB.
// ---------------------------------------------------------------------------
__global__ void __launch_bounds__(256)
k1_vf_swizzle(const float* __restrict__ v, const float* __restrict__ fb0,
              bf16_t* __restrict__ bfile) {
  int gid = blockIdx.x * 256 + threadIdx.x;
  int i = gid >> 6;   // variable index = K of GEMM1
  int n = gid & 63;   // output dim    = N of GEMM1
  const float* vr = v + (size_t)i * D;
  const float* wr = fb0 + (size_t)n * D;
  float acc = 0.f;
#pragma unroll
  for (int e = 0; e < D; ++e) acc += vr[e] * wr[e];
  int ks = i >> 5, kk = i & 31, nt = n >> 4, nn = n & 15;
  size_t idx = (((size_t)ks * 4 + nt) * 32 + frag_lane(kk, nn)) * 16 + frag_half(kk);
  bfile[idx] = (bf16_t)acc;
}

// ---------------------------------------------------------------------------
// K2: c = relu(cm @ vars_fwd + cb), cm = pos - neg streamed from HBM.
// One wave = 16(M) x 64(N) tile, K-loop over all 8192 variables.
// B fragments staged 8 ksteps at a time through LDS (cuts L2 traffic 8x).
// SPILL=true additionally emits the bf16 cm copy for GEMM2 (compile-time,
// so the inner loop has no divergent branch).
// ---------------------------------------------------------------------------
template <bool SPILL>
__global__ void __launch_bounds__(256)
k2_gemm1(const float* __restrict__ pos, const float* __restrict__ neg,
         const bf16_t* __restrict__ bfile, const float* __restrict__ cb,
         bf16_t* __restrict__ cout, bf16_t* __restrict__ cmbf) {
  __shared__ v16bf bsh[8 * 4 * 32];   // 8 ksteps x 4 ntiles x 32 lanes = 32 KB

  int tid  = threadIdx.x;
  int lane = tid & 31;
  int w    = tid >> 5;
  int m0   = (blockIdx.x * 8 + w) * 16;      // 16-row clause block per wave
  int g    = lane >> 4;                      // lane half-group
  int row  = m0 + (lane & 15);
  const float* prow = pos + (size_t)row * NV;
  const float* nrow = neg + (size_t)row * NV;
  bf16_t* crow = SPILL ? (cmbf + (size_t)row * NV) : (bf16_t*)0;
  const v16bf* bfv = (const v16bf*)bfile;

  v8f acc0 = {}, acc1 = {}, acc2 = {}, acc3 = {};

  for (int c = 0; c < (NV / 32) / 8; ++c) {   // 32 chunks of 8 ksteps
    // Stage this chunk's B fragments: load before the barrier so the L2
    // traffic overlaps other waves' previous-chunk WMMAs.
    v16bf t[4];
#pragma unroll
    for (int r = 0; r < 4; ++r)
      t[r] = bfv[(size_t)c * 1024 + r * 256 + tid];
    // Prefetch next chunk's incidence rows.
    if (c + 1 < 32) {
      __builtin_prefetch(prow + (c + 1) * 256 + g * 8, 0, 0);
      __builtin_prefetch(nrow + (c + 1) * 256 + g * 8, 0, 0);
    }
    __syncthreads();
#pragma unroll
    for (int r = 0; r < 4; ++r)
      bsh[r * 256 + tid] = t[r];
    __syncthreads();

#pragma unroll 2
    for (int ksl = 0; ksl < 8; ++ksl) {
      int ks = c * 8 + ksl;
      int k0 = ks * 32 + g * 8;
      // A fragment: two contiguous 8-float runs per lane, pos and neg
      float4 p0 = *(const float4*)(prow + k0);
      float4 p1 = *(const float4*)(prow + k0 + 4);
      float4 p2 = *(const float4*)(prow + k0 + 16);
      float4 p3 = *(const float4*)(prow + k0 + 20);
      float4 q0 = *(const float4*)(nrow + k0);
      float4 q1 = *(const float4*)(nrow + k0 + 4);
      float4 q2 = *(const float4*)(nrow + k0 + 16);
      float4 q3 = *(const float4*)(nrow + k0 + 20);
      v8bf alo = {}, ahi = {};
      alo[0] = (bf16_t)(p0.x - q0.x); alo[1] = (bf16_t)(p0.y - q0.y);
      alo[2] = (bf16_t)(p0.z - q0.z); alo[3] = (bf16_t)(p0.w - q0.w);
      alo[4] = (bf16_t)(p1.x - q1.x); alo[5] = (bf16_t)(p1.y - q1.y);
      alo[6] = (bf16_t)(p1.z - q1.z); alo[7] = (bf16_t)(p1.w - q1.w);
      ahi[0] = (bf16_t)(p2.x - q2.x); ahi[1] = (bf16_t)(p2.y - q2.y);
      ahi[2] = (bf16_t)(p2.z - q2.z); ahi[3] = (bf16_t)(p2.w - q2.w);
      ahi[4] = (bf16_t)(p3.x - q3.x); ahi[5] = (bf16_t)(p3.y - q3.y);
      ahi[6] = (bf16_t)(p3.z - q3.z); ahi[7] = (bf16_t)(p3.w - q3.w);
      if constexpr (SPILL) {   // compile-time: spill bf16 cm for GEMM2
        *(v4u*)(crow + k0)      = __builtin_bit_cast(v4u, alo);
        *(v4u*)(crow + k0 + 16) = __builtin_bit_cast(v4u, ahi);
      }
      v16bf a = __builtin_shufflevector(alo, ahi, 0, 1, 2, 3, 4, 5, 6, 7,
                                        8, 9, 10, 11, 12, 13, 14, 15);

      v16bf b0 = bsh[(ksl * 4 + 0) * 32 + lane];
      v16bf b1 = bsh[(ksl * 4 + 1) * 32 + lane];
      v16bf b2 = bsh[(ksl * 4 + 2) * 32 + lane];
      v16bf b3 = bsh[(ksl * 4 + 3) * 32 + lane];

      acc0 = __builtin_amdgcn_wmma_f32_16x16x32_bf16(false, a, false, b0, (short)0, acc0, false, false);
      acc1 = __builtin_amdgcn_wmma_f32_16x16x32_bf16(false, a, false, b1, (short)0, acc1, false, false);
      acc2 = __builtin_amdgcn_wmma_f32_16x16x32_bf16(false, a, false, b2, (short)0, acc2, false, false);
      acc3 = __builtin_amdgcn_wmma_f32_16x16x32_bf16(false, a, false, b3, (short)0, acc3, false, false);
    }
  }

  // Epilogue: +cb, relu, store bf16 c
  int nb = lane & 15;
#pragma unroll
  for (int j = 0; j < 8; ++j) {
    int m = m0 + j + 8 * g;
    float v0 = fmaxf(acc0[j] + cb[ 0 + nb], 0.f);
    float v1 = fmaxf(acc1[j] + cb[16 + nb], 0.f);
    float v2 = fmaxf(acc2[j] + cb[32 + nb], 0.f);
    float v3 = fmaxf(acc3[j] + cb[48 + nb], 0.f);
    cout[(size_t)m * 64 +  0 + nb] = (bf16_t)v0;
    cout[(size_t)m * 64 + 16 + nb] = (bf16_t)v1;
    cout[(size_t)m * 64 + 32 + nb] = (bf16_t)v2;
    cout[(size_t)m * 64 + 48 + nb] = (bf16_t)v3;
  }
}

// ---------------------------------------------------------------------------
// K3: Y = c @ fb1^T, written as A-fragment-swizzled bf16 file for GEMM2
// (GEMM2 computes nv^T = Y^T @ cm, so Y^T is the A operand, K = clauses).
// ---------------------------------------------------------------------------
__global__ void __launch_bounds__(256)
k3_y_swizzle(const bf16_t* __restrict__ c, const float* __restrict__ fb1,
             bf16_t* __restrict__ afile) {
  int gid = blockIdx.x * 256 + threadIdx.x;
  int k = gid >> 6;   // clause index = K of GEMM2
  int m = gid & 63;   // dim          = M of GEMM2
  const bf16_t* cr = c + (size_t)k * D;
  const float* wr = fb1 + (size_t)m * D;
  float acc = 0.f;
#pragma unroll
  for (int e = 0; e < D; ++e) acc += (float)cr[e] * wr[e];
  int ks = k >> 5, kk = k & 31, mt = m >> 4, mm = m & 15;
  size_t idx = (((size_t)ks * 4 + mt) * 32 + frag_lane(kk, mm)) * 16 + frag_half(kk);
  afile[idx] = (bf16_t)acc;
}

// ---------------------------------------------------------------------------
// K4: nv^T partial = Y^T @ cm over one K segment.
// Block = 64(M=dims) x 32(N=variables) tile. cm tile (128 clauses x 32 vars)
// staged through LDS with a two-phase load->barrier->swizzle-store->barrier
// pattern so global latency overlaps the previous chunk's WMMAs.
// PACKED=true reads the bf16 cm spilled by K2 (0.5 GiB) else f32 pos/neg.
// grid = (NV/32 n-blocks, NSEG k-segments); partial layout [seg][var][dim].
// ---------------------------------------------------------------------------
template <bool PACKED>
__global__ void __launch_bounds__(256)
k4_gemm2(const float* __restrict__ pos, const float* __restrict__ neg,
         const bf16_t* __restrict__ cmbf,
         const bf16_t* __restrict__ afile, float* __restrict__ partial) {
  __shared__ v16bf bfragv[4 * 2 * 32];     // [kl][nt][lane] pre-swizzled, 8 KB
  bf16_t* bstage = (bf16_t*)bfragv;

  int tid  = threadIdx.x;
  int lane = tid & 31;
  int w    = tid >> 5;
  int mt   = w & 3;        // dim tile (4 x 16 = 64 dims)
  int nt   = w >> 2;       // variable tile (2 x 16 = 32 vars)
  int n0   = blockIdx.x * 32;
  int kseg0 = blockIdx.y * (NC / NSEG);

  v8f acc = {};

  for (int chunk = 0; chunk < (NC / NSEG) / 128; ++chunk) {
    int kc0 = kseg0 + chunk * 128;
    bf16_t rv[16];
    // ---- load phase (no LDS): overlaps previous chunk's compute ----
    if constexpr (PACKED) {
#pragma unroll
      for (int it = 0; it < 8; ++it) {
        int e2  = it * 512 + tid * 2;
        int kkl = e2 >> 5;
        int nb  = e2 & 31;
        unsigned int u = *(const unsigned int*)(cmbf + (size_t)(kc0 + kkl) * NV + n0 + nb);
        rv[it * 2]     = __builtin_bit_cast(bf16_t, (unsigned short)(u & 0xffffu));
        rv[it * 2 + 1] = __builtin_bit_cast(bf16_t, (unsigned short)(u >> 16));
      }
    } else {
#pragma unroll
      for (int it = 0; it < 16; ++it) {
        int e   = it * 256 + tid;
        int kkl = e >> 5;
        int nnl = e & 31;
        size_t ga = (size_t)(kc0 + kkl) * NV + n0 + nnl;
        rv[it] = (bf16_t)(pos[ga] - neg[ga]);
      }
    }
    __syncthreads();
    // ---- swizzle-store phase ----
    if constexpr (PACKED) {
#pragma unroll
      for (int it = 0; it < 8; ++it) {
        int e2  = it * 512 + tid * 2;
        int kkl = e2 >> 5;
        int nb  = e2 & 31;
        int kl = kkl >> 5, kk = kkl & 31;
#pragma unroll
        for (int j = 0; j < 2; ++j) {
          int nnl = nb + j;
          int bt = nnl >> 4, nn = nnl & 15;
          bstage[(((kl * 2 + bt) * 32) + frag_lane(kk, nn)) * 16 + frag_half(kk)] = rv[it * 2 + j];
        }
      }
    } else {
#pragma unroll
      for (int it = 0; it < 16; ++it) {
        int e   = it * 256 + tid;
        int kkl = e >> 5;
        int nnl = e & 31;
        int kl = kkl >> 5, kk = kkl & 31, bt = nnl >> 4, nn = nnl & 15;
        bstage[(((kl * 2 + bt) * 32) + frag_lane(kk, nn)) * 16 + frag_half(kk)] = rv[it];
      }
    }
    __syncthreads();
    // ---- compute phase: 4 ksteps ----
#pragma unroll
    for (int kl = 0; kl < 4; ++kl) {
      int ks = (kc0 >> 5) + kl;  // global 32-wide K step
      v16bf a = ((const v16bf*)afile)[((size_t)ks * 4 + mt) * 32 + lane];
      v16bf b = bfragv[(kl * 2 + nt) * 32 + lane];
      acc = __builtin_amdgcn_wmma_f32_16x16x32_bf16(false, a, false, b, (short)0, acc, false, false);
    }
  }

  // Write f32 partial[seg][var][dim]: two contiguous float4 per lane.
  int n = n0 + nt * 16 + (lane & 15);
  int g = lane >> 4;
  size_t base = ((size_t)blockIdx.y * NV + n) * 64 + mt * 16 + 8 * g;
  float4 f0, f1;
  f0.x = acc[0]; f0.y = acc[1]; f0.z = acc[2]; f0.w = acc[3];
  f1.x = acc[4]; f1.y = acc[5]; f1.z = acc[6]; f1.w = acc[7];
  *(float4*)(partial + base)     = f0;
  *(float4*)(partial + base + 4) = f1;
}

// ---------------------------------------------------------------------------
// K5: reduce K-split partials, relu(+vb), ground combiner, GRU, final output.
// Block = 4 variables x 64 dims, phased via LDS. partial is [seg][var][dim]
// so the reduce is fully coalesced.
// ---------------------------------------------------------------------------
__device__ __forceinline__ float sigmoidf(float x) { return 1.f / (1.f + __expf(-x)); }

__global__ void __launch_bounds__(256)
k5_finish(const float* __restrict__ partial, const float* __restrict__ vb,
          const float* __restrict__ gvars, const float* __restrict__ Wg,
          const float* __restrict__ bg, const float* __restrict__ prev_all,
          const float* __restrict__ Wz, const float* __restrict__ Uz, const float* __restrict__ bz,
          const float* __restrict__ Wr, const float* __restrict__ Ur, const float* __restrict__ br,
          const float* __restrict__ Wh, const float* __restrict__ Uh, const float* __restrict__ bh,
          float* __restrict__ out) {
  __shared__ float emb[4][64];
  __shared__ float prv[4][64];
  __shared__ float rp[4][64];

  int tid = threadIdx.x;
  int il  = tid >> 6;       // local variable 0..3
  int d   = tid & 63;       // dim
  int i   = blockIdx.x * 4 + il;

  float nv = 0.f;
#pragma unroll
  for (int s = 0; s < NSEG; ++s) nv += partial[((size_t)s * NV + i) * 64 + d];
  emb[il][d] = fmaxf(nv + vb[d], 0.f);
  prv[il][d] = prev_all[(size_t)i * 64 + d];
  __syncthreads();

  // ground combiner: tanh(Wg @ [ground, relu_emb] + bg)
  float accg = bg[d];
  const float* wgr = Wg + d * (GD + D);
#pragma unroll
  for (int g = 0; g < GD; ++g) accg += wgr[g] * gvars[(size_t)i * GD + g];
#pragma unroll
  for (int e = 0; e < D; ++e) accg += wgr[GD + e] * emb[il][e];
  float vemb = tanhf(accg);
  __syncthreads();
  emb[il][d] = vemb;
  __syncthreads();

  // GRU gates
  float az = bz[d], ar = br[d];
#pragma unroll
  for (int e = 0; e < D; ++e) {
    az += Wz[d * 64 + e] * emb[il][e] + Uz[d * 64 + e] * prv[il][e];
    ar += Wr[d * 64 + e] * emb[il][e] + Ur[d * 64 + e] * prv[il][e];
  }
  float z = sigmoidf(az);
  float r = sigmoidf(ar);
  rp[il][d] = r * prv[il][d];
  __syncthreads();

  float ah = bh[d];
#pragma unroll
  for (int e = 0; e < D; ++e)
    ah += Wh[d * 64 + e] * emb[il][e] + Uh[d * 64 + e] * rp[il][e];
  float h = tanhf(ah);

  out[(size_t)i * 64 + d] = (1.f - z) * prv[il][d] + z * h;
}

// ---------------------------------------------------------------------------
extern "C" void kernel_launch(void* const* d_in, const int* in_sizes, int n_in,
                              void* d_out, int out_size, void* d_ws, size_t ws_size,
                              hipStream_t stream) {
  (void)in_sizes; (void)n_in; (void)out_size;
  const float* variables = (const float*)d_in[0];
  const float* ground    = (const float*)d_in[1];
  const float* cpos      = (const float*)d_in[2];
  const float* cneg      = (const float*)d_in[3];
  const float* fb        = (const float*)d_in[6];   // [2][64][64]
  const float* vb        = (const float*)d_in[7];
  const float* cb        = (const float*)d_in[8];
  const float* Wg        = (const float*)d_in[9];
  const float* bg        = (const float*)d_in[10];
  const float* Wz = (const float*)d_in[11];
  const float* Uz = (const float*)d_in[12];
  const float* bz = (const float*)d_in[13];
  const float* Wr = (const float*)d_in[14];
  const float* Ur = (const float*)d_in[15];
  const float* br = (const float*)d_in[16];
  const float* Wh = (const float*)d_in[17];
  const float* Uh = (const float*)d_in[18];
  const float* bh = (const float*)d_in[19];

  char* ws = (char*)d_ws;
  size_t offB = 0;                                   // 1 MB  (bfile)
  size_t offC = offB + (size_t)NV * 64 * 2;          // 4 MB  (c bf16)
  size_t offA = offC + (size_t)NC * 64 * 2;          // 4 MB  (afile)
  size_t offP = offA + (size_t)NC * 64 * 2;          // 8 MB  (partials)
  size_t offM = offP + (size_t)NSEG * 64 * NV * 4;   // 512 MB (packed cm, optional)
  size_t need = offM + (size_t)NC * NV * 2;

  bf16_t* bfile   = (bf16_t*)(ws + offB);
  bf16_t* cbuf    = (bf16_t*)(ws + offC);
  bf16_t* afile   = (bf16_t*)(ws + offA);
  float*  partial = (float*)(ws + offP);
  bf16_t* cmbf    = (ws_size >= need) ? (bf16_t*)(ws + offM) : (bf16_t*)0;
  float* out = (float*)d_out;

  k1_vf_swizzle<<<dim3(NV * 64 / 256), dim3(256), 0, stream>>>(variables, fb, bfile);
  if (cmbf)
    k2_gemm1<true><<<dim3(NC / 128), dim3(256), 0, stream>>>(cpos, cneg, bfile, cb, cbuf, cmbf);
  else
    k2_gemm1<false><<<dim3(NC / 128), dim3(256), 0, stream>>>(cpos, cneg, bfile, cb, cbuf, (bf16_t*)0);
  k3_y_swizzle<<<dim3(NC * 64 / 256), dim3(256), 0, stream>>>(cbuf, fb + 64 * 64, afile);
  if (cmbf)
    k4_gemm2<true><<<dim3(NV / 32, NSEG), dim3(256), 0, stream>>>(cpos, cneg, cmbf, afile, partial);
  else
    k4_gemm2<false><<<dim3(NV / 32, NSEG), dim3(256), 0, stream>>>(cpos, cneg, (bf16_t*)0, afile, partial);
  k5_finish<<<dim3(NV / 4), dim3(256), 0, stream>>>(partial, vb, ground, Wg, bg, variables,
                                                    Wz, Uz, bz, Wr, Ur, br, Wh, Uh, bh, out);
}
